// EmpiricalDFM_5617817224099
// MI455X (gfx1250) — compile-verified, matrix operand order checked by reading, not run on hardware
//
#include <hip/hip_runtime.h>
#include <hip/hip_bf16.h>
#include <math.h>

#define NUM_TOKENS 50265
#define CTX 384

typedef int v8i_ __attribute__((ext_vector_type(8)));

// -------------------------------------------------------------------------
// Kernel 1: match_count via V_WMMA_I32_16X16X64_IU8.
// One wave (32 lanes) handles one tile of 16 dataset rows for one batch b.
// A-matrix: 0/1 equality bytes (lane m & m+16 jointly cover 64 positions of
// row m). B-matrix: all-ones bytes -> D[m][*] = row popcount. K=384 = 6 x 64.
// -------------------------------------------------------------------------
__global__ void match_wmma_kernel(const int* __restrict__ dataset,
                                  const int* __restrict__ input,
                                  int* __restrict__ mc,
                                  int n_ds, int bs) {
    __shared__ int s_in[8 * CTX];               // input rows cached in LDS
    const int tid = threadIdx.x;
    for (int i = tid; i < bs * CTX; i += blockDim.x) s_in[i] = input[i];
    __syncthreads();

    const int b    = tid >> 5;                  // wave id == batch index
    const int lane = tid & 31;
    const int half = lane >> 4;                 // which 32-position half
    const int m    = lane & 15;                 // A-matrix row within tile
    const int tile = blockIdx.x;
    const int row  = tile * 16 + m;

    const int* __restrict__ drow = dataset + (size_t)row * CTX;
    const int* __restrict__ irow = s_in + b * CTX;

    v8i_ acc = {};
    v8i_ ones;
#pragma unroll
    for (int j = 0; j < 8; ++j) ones[j] = 0x01010101;

#pragma unroll
    for (int chunk = 0; chunk < 6; ++chunk) {
        const int base_s = chunk * 64 + half * 32;
        if (chunk < 5)
            __builtin_prefetch(drow + base_s + 64, 0, 3);   // global_prefetch_b8
        v8i_ a;
#pragma unroll
        for (int j = 0; j < 8; ++j) {
            const int4 dv = *(const int4*)(drow + base_s + j * 4);
            unsigned p = 0;
            p |= (dv.x == irow[base_s + j * 4 + 0]) ? 1u : 0u;
            p |= (dv.y == irow[base_s + j * 4 + 1]) ? (1u << 8) : 0u;
            p |= (dv.z == irow[base_s + j * 4 + 2]) ? (1u << 16) : 0u;
            p |= (dv.w == irow[base_s + j * 4 + 3]) ? (1u << 24) : 0u;
            a[j] = (int)p;
        }
        // D = A(16x64 u8) * ones(64x16 u8) + C : every column = row popcount
        acc = __builtin_amdgcn_wmma_i32_16x16x64_iu8(false, a, false, ones,
                                                     acc, false, false);
    }

    // Column N=0 lives in lane 0 (M=0..7 in vgprs 0..7) and lane 16 (M=8..15)
    if (m == 0) {
        int* dst = mc + (size_t)b * n_ds + tile * 16 + half * 8;
#pragma unroll
        for (int j = 0; j < 8; ++j) dst[j] = acc[j];
    }
}

// -------------------------------------------------------------------------
// Kernel 2: w[b,n] = softmax_n( mc*log_a + (C-mc)*log_b ). One block per b.
// -------------------------------------------------------------------------
__global__ void weights_kernel(const int* __restrict__ mc,
                               const float* __restrict__ t,
                               float* __restrict__ w, int n_ds) {
    __shared__ float red[256];
    const int b   = blockIdx.x;
    const int tid = threadIdx.x;
    const float tb = t[b];
    const float la = logf(tb + (1.0f - tb) / (float)NUM_TOKENS);
    const float lb = logf((1.0f - tb) / (float)NUM_TOKENS);
    const int* __restrict__ mcb = mc + (size_t)b * n_ds;

    float mx = -3.4e38f;
    for (int n = tid; n < n_ds; n += blockDim.x) {
        const float c = (float)mcb[n];
        mx = fmaxf(mx, c * la + ((float)CTX - c) * lb);
    }
    red[tid] = mx; __syncthreads();
    for (int s = blockDim.x >> 1; s > 0; s >>= 1) {
        if (tid < s) red[tid] = fmaxf(red[tid], red[tid + s]);
        __syncthreads();
    }
    mx = red[0]; __syncthreads();

    float sum = 0.0f;
    for (int n = tid; n < n_ds; n += blockDim.x) {
        const float c = (float)mcb[n];
        sum += expf(c * la + ((float)CTX - c) * lb - mx);
    }
    red[tid] = sum; __syncthreads();
    for (int s = blockDim.x >> 1; s > 0; s >>= 1) {
        if (tid < s) red[tid] += red[tid + s];
        __syncthreads();
    }
    const float inv = 1.0f / red[0];

    float* __restrict__ wb = w + (size_t)b * n_ds;
    for (int n = tid; n < n_ds; n += blockDim.x) {
        const float c = (float)mcb[n];
        wb[n] = expf(c * la + ((float)CTX - c) * lb - mx) * inv;
    }
}

// -------------------------------------------------------------------------
// Kernel 3: zero-fill 309 MB output with float4 stores (HBM-bound, ~13us).
// -------------------------------------------------------------------------
__global__ void zero_kernel(float* __restrict__ out, long long n) {
    const long long n4     = n >> 2;
    const long long stride = (long long)gridDim.x * blockDim.x;
    long long i = (long long)blockIdx.x * blockDim.x + threadIdx.x;
    float4* __restrict__ o4 = (float4*)out;
    const float4 z = {0.0f, 0.0f, 0.0f, 0.0f};
    for (long long k = i; k < n4; k += stride) o4[k] = z;
    const long long tail = n - (n4 << 2);
    if (i < tail) out[(n4 << 2) + i] = 0.0f;
}

// -------------------------------------------------------------------------
// Kernel 4: scatter  out[b, s, dataset[n,s]] += w[b,n]
// One thread per (n,s) pair; 4 atomics each (no-return -> STOREcnt path).
// -------------------------------------------------------------------------
__global__ void scatter_kernel(const int* __restrict__ dataset,
                               const float* __restrict__ w,
                               float* __restrict__ out,
                               int n_ds, int bs) {
    const long long total = (long long)n_ds * CTX;
    const long long idx = (long long)blockIdx.x * blockDim.x + threadIdx.x;
    if (idx >= total) return;
    const int n   = (int)(idx / CTX);
    const int s   = (int)(idx % CTX);
    const int tok = dataset[idx];
    for (int b = 0; b < bs; ++b) {
        const float wv = w[(size_t)b * n_ds + n];
        atomicAdd(out + ((size_t)b * CTX + s) * (size_t)NUM_TOKENS + tok, wv);
    }
}

extern "C" void kernel_launch(void* const* d_in, const int* in_sizes, int n_in,
                              void* d_out, int out_size, void* d_ws, size_t ws_size,
                              hipStream_t stream) {
    const int*   input   = (const int*)d_in[0];    // (BS, CTX)
    const int*   dataset = (const int*)d_in[1];    // (N, CTX)
    const float* t       = (const float*)d_in[2];  // (BS,)
    float*       out     = (float*)d_out;          // (BS, CTX, NUM_TOKENS)

    const int bs   = in_sizes[0] / CTX;            // 4
    const int n_ds = in_sizes[1] / CTX;            // 2048

    // workspace: match counts (int) then weights (float), 256B aligned
    int*   mc = (int*)d_ws;
    size_t mc_bytes = ((size_t)bs * n_ds * sizeof(int) + 255) & ~(size_t)255;
    float* w  = (float*)((char*)d_ws + mc_bytes);

    // 1) WMMA IU8 match counts: one wave per (16-row tile, b)
    const int tiles = n_ds / 16;
    match_wmma_kernel<<<tiles, 32 * bs, 0, stream>>>(dataset, input, mc, n_ds, bs);

    // 2) softmax weights, one block per b
    weights_kernel<<<bs, 256, 0, stream>>>(mc, t, w, n_ds);

    // 3) zero the 309MB output (grid-stride float4)
    zero_kernel<<<8192, 256, 0, stream>>>(out, (long long)out_size);

    // 4) scatter-add weights
    const long long pairs = (long long)n_ds * CTX;
    const int blocks = (int)((pairs + 255) / 256);
    scatter_kernel<<<blocks, 256, 0, stream>>>(dataset, w, out, n_ds, bs);
}
